// MLP_59296318488700
// MI455X (gfx1250) — matrix operand order repaired and checked by reading, not compiled, not run on hardware
//
#include <hip/hip_runtime.h>
#include <hip/hip_bf16.h>
#include <hip/hip_fp16.h>

// ---------------------------------------------------------------------------
// MoE (top-2 of 16 experts + shared), T=2048, h=2048, i=1024, f32 I/O.
// router -> gather lists -> WMMA SwiGLU (f16 in, f32 accum) -> WMMA down-proj
// LDS tiles in *fragment order*: each lane's v16h operand is one aligned 32B
// contiguous LDS read. A-tiles arrive via GLOBAL_LOAD_ASYNC_TO_LDS_B128
// (ASYNCcnt + s_wait_asynccnt).
// ---------------------------------------------------------------------------

typedef __attribute__((ext_vector_type(16))) _Float16 v16h;
typedef __attribute__((ext_vector_type(2)))  _Float16 v2h;
typedef __attribute__((ext_vector_type(2)))  __fp16   v2fp;   // cvt_pkrtz result
typedef __attribute__((ext_vector_type(8)))  float    v8f;

#define TM 16   // token tile (M)
#define TN 64   // column tile (N): 4 waves x 16
#define TK 32   // K per WMMA step
#define BLK 128 // 4 waves

#if __has_builtin(__builtin_amdgcn_global_load_async_to_lds_b128) && \
    __has_builtin(__builtin_amdgcn_s_wait_asynccnt)
#define HAVE_ASYNC_LDS 1
typedef int b128_t __attribute__((vector_size(16)));
typedef __attribute__((address_space(1))) b128_t glb_b128;
typedef __attribute__((address_space(3))) b128_t lds_b128;
#define AS_GLOBAL(p) ((glb_b128*)(uintptr_t)(p))
#define AS_LDS(p)    ((lds_b128*)(unsigned)(uintptr_t)(p))
#else
#define HAVE_ASYNC_LDS 0
#endif

__device__ __forceinline__ float fast_rcp(float x) {
#if __has_builtin(__builtin_amdgcn_rcpf)
  return __builtin_amdgcn_rcpf(x);   // v_rcp_f32
#else
  return 1.f / x;
#endif
}

// Fragment-order helpers (CDNA5 ISA §7.12.2):
// A 16x32 f16: lane (h*16+row), element q -> A[row][ q<8 ? 8h+q : 16+8h+(q-8) ]
//   inverse: hi=k>>4, rem=k&15, h=rem>>3, q=(rem&7)+8*hi, lane=h*16+row
// B 32x16 f16 (per 16-col wave slice): lane (h*16+col), element q -> B[ q<8 ? 8h+q : 16+8h+(q-8) ][col]

// ---------------------------------------------------------------------------
// Kernel 1: router (VALU; 67 MFLOP total — noise vs the FFNs)
// ---------------------------------------------------------------------------
__global__ void router_kernel(const float* __restrict__ x,
                              const float* __restrict__ Wr,
                              const float* __restrict__ bias,
                              int*  __restrict__ counts,
                              int*  __restrict__ tokslot,
                              float* __restrict__ wlist,
                              int T, int H) {
  const int lane = threadIdx.x & 31;
  const int wv   = threadIdx.x >> 5;
  const int t    = blockIdx.x * (blockDim.x >> 5) + wv;
  if (t >= T) return;

  float acc[16];
#pragma unroll
  for (int e = 0; e < 16; ++e) acc[e] = 0.f;
  for (int k = lane; k < H; k += 32) {
    const float xv = x[(size_t)t * H + k];
#pragma unroll
    for (int e = 0; e < 16; ++e) acc[e] = fmaf(xv, Wr[e * H + k], acc[e]);
  }
#pragma unroll
  for (int e = 0; e < 16; ++e)
#pragma unroll
    for (int off = 16; off > 0; off >>= 1)
      acc[e] += __shfl_xor(acc[e], off, 32);

  if (lane == 0) {
    float l[16], p[16], m = -3.4e38f;
#pragma unroll
    for (int e = 0; e < 16; ++e) { l[e] = acc[e] + bias[e]; m = fmaxf(m, l[e]); }
    float s = 0.f;
#pragma unroll
    for (int e = 0; e < 16; ++e) { p[e] = __expf(l[e] - m); s += p[e]; }
    int i0 = 0;
#pragma unroll
    for (int e = 1; e < 16; ++e) if (p[e] > p[i0]) i0 = e;
    int i1 = (i0 == 0) ? 1 : 0;
#pragma unroll
    for (int e = 0; e < 16; ++e) if (e != i0 && p[e] > p[i1]) i1 = e;
    const float rs = fast_rcp(s);
    const float p0 = p[i0] * rs, p1 = p[i1] * rs;
    const float rw = fast_rcp(p0 + p1 + 1e-9f);
    int s0 = atomicAdd(&counts[i0], 1);
    tokslot[i0 * T + s0] = t * 2 + 0;
    wlist  [i0 * T + s0] = p0 * rw;
    int s1 = atomicAdd(&counts[i1], 1);
    tokslot[i1 * T + s1] = t * 2 + 1;
    wlist  [i1 * T + s1] = p1 * rw;
  }
}

// ---------------------------------------------------------------------------
// Kernel 2: gate/up + SwiGLU epilogue (routing weight folded into act).
// grid = (I/TN, T/TM, E+1); z==E is the shared expert over all tokens.
// ---------------------------------------------------------------------------
__global__ void moe_gu_kernel(const float* __restrict__ x,
                              const float* __restrict__ Wg,
                              const float* __restrict__ Wu,
                              const float* __restrict__ sWg,
                              const float* __restrict__ sWu,
                              const int*  __restrict__ counts,
                              const int*  __restrict__ tokslot,
                              const float* __restrict__ wlist,
                              _Float16* __restrict__ act,
                              int T, int H, int I) {
  const int e  = blockIdx.z;
  const int m0 = blockIdx.y * TM;
  const int n0 = blockIdx.x * TN;
  const int E  = gridDim.z - 1;

  const int cnt = (e < E) ? counts[e] : T;
  const int nt  = cnt < T ? cnt : T;
  if (m0 >= nt) return;

  __shared__ int   sTok[TM];
  __shared__ int   sRow[TM];
  __shared__ float sW[TM];
  __shared__ __align__(32) float    AsF[32 * 16];      // A tile, fragment order (f32)
  __shared__ __align__(32) _Float16 BgF[4 * 32 * 16];  // B tiles, fragment order (f16)
  __shared__ __align__(32) _Float16 BuF[4 * 32 * 16];

  const int tid  = threadIdx.x;
  const int lane = tid & 31;
  const int wv   = tid >> 5;

  if (tid < TM) {
    int tok = 0, arow = 3 * T; float w = 0.f;
    if (m0 + tid < nt) {
      if (e < E) {
        const int ent = tokslot[e * T + m0 + tid];
        tok = ent >> 1; arow = tok * 3 + (ent & 1);
        w   = wlist[e * T + m0 + tid];
      } else { tok = m0 + tid; arow = tok * 3 + 2; w = 1.f; }
    }
    sTok[tid] = tok; sRow[tid] = arow; sW[tid] = w;
  }
  __syncthreads();

  const float* bg = (e < E) ? (Wg + (size_t)e * H * I) : sWg;
  const float* bu = (e < E) ? (Wu + (size_t)e * H * I) : sWu;

  const int    hA   = lane >> 4;
  const size_t xRow = (size_t)sTok[lane & 15] * H;  // per-lane token row

  v8f cg = {}; v8f cu = {};

  for (int k0 = 0; k0 < H; k0 += TK) {
    // ---- A tile: x rows -> fragment-order f32 LDS -------------------------
#if HAVE_ASYNC_LDS
    {  // wave wv fetches 16B chunk wv of every lane's fragment run
      const int kb = (wv < 2) ? (8 * hA + 4 * wv) : (16 + 8 * hA + 4 * (wv - 2));
      __builtin_amdgcn_global_load_async_to_lds_b128(
          AS_GLOBAL(x + xRow + k0 + kb), AS_LDS(&AsF[(lane << 4) + 4 * wv]), 0, 0);
    }
#else
#pragma unroll
    for (int p = tid; p < TM * TK; p += BLK) {
      const int r = p >> 5, k = p & 31;
      const int hi = k >> 4, rem = k & 15, h = rem >> 3, q = (rem & 7) + 8 * hi;
      AsF[((h * 16 + r) << 4) + q] = x[(size_t)sTok[r] * H + k0 + k];
    }
#endif
    // ---- B tiles: weight rows -> fragment-order f16 (pk-converted) -------
#pragma unroll
    for (int p = tid; p < (TK / 2) * TN; p += BLK) {
      const int kkp = p >> 6, c = p & 63;          // kk pair, column
      const int kk = kkp * 2;
      const int l  = (((kk >> 3) & 1) << 4) + (c & 15);
      const int q  = (kk & 7) + ((kk >> 4) << 3);   // even
      const int w  = c >> 4;
      const size_t go = (size_t)(k0 + kk) * I + n0 + c;
      const v2fp pg = __builtin_amdgcn_cvt_pkrtz(bg[go], bg[go + I]);
      const v2fp pu = __builtin_amdgcn_cvt_pkrtz(bu[go], bu[go + I]);
      *(v2fp*)&BgF[(((w << 5) + l) << 4) + q] = pg;
      *(v2fp*)&BuF[(((w << 5) + l) << 4) + q] = pu;
    }
    if (k0 + TK < H) {  // hint next weight tile toward L2 (global_prefetch_b8)
      __builtin_prefetch(&bg[(size_t)(k0 + TK) * I + n0 + (tid & 63)], 0, 1);
      __builtin_prefetch(&bu[(size_t)(k0 + TK) * I + n0 + (tid & 63)], 0, 1);
    }
#if HAVE_ASYNC_LDS
    __builtin_amdgcn_s_wait_asynccnt(0);
#endif
    __syncthreads();

    // ---- fragments: one aligned 32B run per lane --------------------------
    v16h a;
    {
      const float* ap = &AsF[lane << 4];
#pragma unroll
      for (int v = 0; v < 8; ++v) {
        const v2fp pk = __builtin_amdgcn_cvt_pkrtz(ap[2 * v], ap[2 * v + 1]);
        a[2 * v]     = (_Float16)pk.x;
        a[2 * v + 1] = (_Float16)pk.y;
      }
    }
    const v16h bgf = *(const v16h*)&BgF[((wv << 5) + lane) << 4];
    const v16h buf = *(const v16h*)&BuF[((wv << 5) + lane) << 4];
    cg = __builtin_amdgcn_wmma_f32_16x16x32_f16(false, a, false, bgf, (short)0, cg, false, false);
    cu = __builtin_amdgcn_wmma_f32_16x16x32_f16(false, a, false, buf, (short)0, cu, false, false);
    __syncthreads();
  }

  // ---- epilogue: act = silu(g) * u * w  (C layout: M = v + 8*(lane>>4)) ---
  const int col = n0 + wv * 16 + (lane & 15);
  const int rb  = (lane >> 4) * 8;
#pragma unroll
  for (int v = 0; v < 8; ++v) {
    const int r = rb + v;
    const float g = cg[v], u = cu[v];
    const float silu = g * fast_rcp(1.f + __expf(-g));
    act[(size_t)sRow[r] * I + col] = (_Float16)(silu * u * sW[r]);
  }
}

// ---------------------------------------------------------------------------
// Kernel 3: down projection act(f16) . Wd, atomic combine into out.
// ---------------------------------------------------------------------------
__global__ void moe_down_kernel(const _Float16* __restrict__ act,
                                const float* __restrict__ Wd,
                                const float* __restrict__ sWd,
                                const int*  __restrict__ counts,
                                const int*  __restrict__ tokslot,
                                float* __restrict__ out,
                                int T, int H, int I) {
  const int e  = blockIdx.z;
  const int m0 = blockIdx.y * TM;
  const int n0 = blockIdx.x * TN;
  const int E  = gridDim.z - 1;

  const int cnt = (e < E) ? counts[e] : T;
  const int nt  = cnt < T ? cnt : T;
  if (m0 >= nt) return;

  __shared__ int sTok[TM];
  __shared__ int sRow[TM];
  __shared__ __align__(32) _Float16 AsF[32 * 16];      // fragment-order f16
  __shared__ __align__(32) _Float16 BdF[4 * 32 * 16];

  const int tid  = threadIdx.x;
  const int lane = tid & 31;
  const int wv   = tid >> 5;

  if (tid < TM) {
    int tok = 0, arow = 3 * T;  // dummy zero row for padded slots
    if (m0 + tid < nt) {
      if (e < E) {
        const int ent = tokslot[e * T + m0 + tid];
        tok = ent >> 1; arow = tok * 3 + (ent & 1);
      } else { tok = m0 + tid; arow = tok * 3 + 2; }
    }
    sTok[tid] = tok; sRow[tid] = arow;
  }
  __syncthreads();

  const float* bd = (e < E) ? (Wd + (size_t)e * I * H) : sWd;

  const int    hA   = lane >> 4;
  const size_t aRow = (size_t)sRow[lane & 15] * I;

  v8f c = {};
  for (int k0 = 0; k0 < I; k0 += TK) {
    // ---- A tile (already f16): two contiguous 16B chunks per lane ---------
#if HAVE_ASYNC_LDS
    if (wv < 2) {
      const int kb = (wv == 0) ? (8 * hA) : (16 + 8 * hA);
      __builtin_amdgcn_global_load_async_to_lds_b128(
          AS_GLOBAL(act + aRow + k0 + kb), AS_LDS(&AsF[(lane << 4) + 8 * wv]), 0, 0);
    }
#else
#pragma unroll
    for (int p = tid; p < TM * TK / 2; p += BLK) {
      const int r = p >> 4, k = (p & 15) * 2;
      const int hi = k >> 4, rem = k & 15, h = rem >> 3, q = (rem & 7) + 8 * hi;
      *(v2h*)&AsF[((h * 16 + r) << 4) + q] =
          *(const v2h*)&act[(size_t)sRow[r] * I + k0 + k];
    }
#endif
    // ---- B tile ------------------------------------------------------------
#pragma unroll
    for (int p = tid; p < (TK / 2) * TN; p += BLK) {
      const int kkp = p >> 6, cc = p & 63;
      const int kk = kkp * 2;
      const int l  = (((kk >> 3) & 1) << 4) + (cc & 15);
      const int q  = (kk & 7) + ((kk >> 4) << 3);
      const int w  = cc >> 4;
      const size_t go = (size_t)(k0 + kk) * H + n0 + cc;
      *(v2fp*)&BdF[(((w << 5) + l) << 4) + q] =
          __builtin_amdgcn_cvt_pkrtz(bd[go], bd[go + H]);
    }
    if (k0 + TK < I)
      __builtin_prefetch(&bd[(size_t)(k0 + TK) * H + n0 + (tid & 63)], 0, 1);
#if HAVE_ASYNC_LDS
    __builtin_amdgcn_s_wait_asynccnt(0);
#endif
    __syncthreads();

    const v16h a = *(const v16h*)&AsF[lane << 4];
    const v16h b = *(const v16h*)&BdF[((wv << 5) + lane) << 4];
    c = __builtin_amdgcn_wmma_f32_16x16x32_f16(false, a, false, b, (short)0, c, false, false);
    __syncthreads();
  }

  const int col = n0 + wv * 16 + (lane & 15);
  const int rb  = (lane >> 4) * 8;
#pragma unroll
  for (int v = 0; v < 8; ++v)
    unsafeAtomicAdd(&out[(size_t)sTok[rb + v] * H + col], c[v]);
}

// ---------------------------------------------------------------------------
// Host launcher (graph-capture safe).
// ---------------------------------------------------------------------------
extern "C" void kernel_launch(void* const* d_in, const int* in_sizes, int n_in,
                              void* d_out, int out_size, void* d_ws, size_t ws_size,
                              hipStream_t stream) {
  const float* x    = (const float*)d_in[0];
  const float* Wr   = (const float*)d_in[1];
  const float* bias = (const float*)d_in[2];
  const float* Wg   = (const float*)d_in[3];
  const float* Wu   = (const float*)d_in[4];
  const float* Wd   = (const float*)d_in[5];
  const float* sWg  = (const float*)d_in[6];
  const float* sWu  = (const float*)d_in[7];
  const float* sWd  = (const float*)d_in[8];
  float* out = (float*)d_out;

  const int E = in_sizes[2];       // 16
  const int H = in_sizes[1] / E;   // 2048
  const int T = in_sizes[0] / H;   // 2048
  const int I = in_sizes[6] / H;   // 1024

  char* ws = (char*)d_ws;
  int*   counts  = (int*)ws;
  int*   tokslot = (int*)(ws + 256);
  float* wlist   = (float*)(ws + 256 + sizeof(int) * (size_t)E * T);
  size_t actOff  = 256 + (sizeof(int) + sizeof(float)) * (size_t)E * T;
  actOff = (actOff + 255) & ~(size_t)255;
  _Float16* act = (_Float16*)(ws + actOff);   // (3T+1) x I f16

  (void)hipMemsetAsync(counts, 0, 256, stream);
  (void)hipMemsetAsync(out, 0, sizeof(float) * (size_t)out_size, stream);

  router_kernel<<<dim3((T + 7) / 8), dim3(256), 0, stream>>>(
      x, Wr, bias, counts, tokslot, wlist, T, H);

  moe_gu_kernel<<<dim3(I / TN, T / TM, E + 1), dim3(BLK), 0, stream>>>(
      x, Wg, Wu, sWg, sWu, counts, tokslot, wlist, act, T, H, I);

  moe_down_kernel<<<dim3(H / TN, T / TM, E + 1), dim3(BLK), 0, stream>>>(
      act, Wd, sWd, counts, tokslot, out, T, H, I);
}
